// STGEmbMamba_77653008711871
// MI455X (gfx1250) — matrix operand order repaired and checked by reading, not compiled, not run on hardware
//
#include <hip/hip_runtime.h>
#include <hip/hip_bf16.h>
#include <stdint.h>

#define BATCH   4
#define TLEN    288
#define NNODE   207
#define PATCHK  12
#define PNUM    24          // TLEN/PATCHK
#define DMODEL  256
#define DSTATE  16
#define DCONV   4
#define DINNER  512
#define DTRANK  16
#define NLAYERS 2
#define PREDLEN 12
#define SEQL    (NNODE*PNUM)   // 4968
#define BL      (BATCH*SEQL)   // 19872
#define XDBLW   (DTRANK + 2*DSTATE)  // 48

// chunked scan: SEQL = NCH * CLEN (92 * 54 = 4968)
#define NCH     92
#define CLEN    54
#define NSTATES (BATCH*DINNER*DSTATE)   // 32768

typedef __attribute__((ext_vector_type(16))) _Float16 v16h;
typedef __attribute__((ext_vector_type(8)))  _Float16 v8h;
typedef __attribute__((ext_vector_type(8)))  float    v8f;

__device__ __forceinline__ float sigmoidf_(float x){ return 1.f/(1.f+__expf(-x)); }

// ---------------- f32 -> f16 weight conversion ----------------
__global__ void f32_to_f16_kernel(const float* __restrict__ in, _Float16* __restrict__ out, int n){
  int i = blockIdx.x*blockDim.x + threadIdx.x;
  if (i < n) out[i] = (_Float16)in[i];
}

// ---------------- patch embedding ----------------
// seq[b, s*256+d] with flat f = n*6144 + c*24 + p ; emb = sum_k inputs[b, p*12+k, n]*pw[c,k] + pb[c]
__global__ void patch_embed_kernel(const float* __restrict__ x, const float* __restrict__ pw,
                                   const float* __restrict__ pb, float* __restrict__ seq,
                                   _Float16* __restrict__ seqh){
  int i = blockIdx.x*blockDim.x + threadIdx.x;
  const int per_b = SEQL*DMODEL;
  if (i >= BATCH*per_b) return;
  int b   = i / per_b;
  int f   = i % per_b;
  int n   = f / (DMODEL*PNUM);
  int rem = f % (DMODEL*PNUM);
  int c   = rem / PNUM;
  int p   = rem % PNUM;
  const float* xin = x + (size_t)b*TLEN*NNODE + (size_t)(p*PATCHK)*NNODE + n;
  const float* w   = pw + c*PATCHK;
  float acc = pb[c];
  #pragma unroll
  for (int k=0;k<PATCHK;k++) acc += xin[(size_t)k*NNODE]*w[k];
  seq[i]  = acc;
  seqh[i] = (_Float16)acc;
}

// ---------------- WMMA GEMM: C[m,n] = sum_k A[m,k]*W[n,k] ----------------
// A: MxK f16 row-major, W: NxK f16 row-major, C: MxN f32.
// One wave = 16 x (NSUB*16) tile. K and NSUB compile-time; M, N are runtime args.
template<int K, int NSUB, bool WRITE_H>
__global__ void __launch_bounds__(256)
gemm_wmma_f16(const _Float16* __restrict__ A, const _Float16* __restrict__ W,
              float* __restrict__ C, _Float16* __restrict__ Ch,
              int M, int N){
  int wave = blockIdx.x*(blockDim.x>>5) + (threadIdx.x>>5);
  int lane = threadIdx.x & 31;
  int mtiles  = M >> 4;
  int ngroups = N / (NSUB*16);
  if (wave >= mtiles*ngroups) return;
  int mt = wave % mtiles;           // consecutive waves share the W tile group
  int ng = wave / mtiles;
  int n0 = ng * (NSUB*16);
  int l16 = lane & 15;
  int g   = lane >> 4;

  v8f acc[NSUB];
  #pragma unroll
  for (int t=0;t<NSUB;t++) acc[t] = (v8f){};

  const _Float16* Arow = A + (size_t)((mt<<4)+l16)*K + g*16;  // 16 contiguous per chunk
  const _Float16* Wb   = W + (size_t)(n0 + l16)*K + g*16;     // 16 contiguous per chunk

  #pragma unroll 2
  for (int k0 = 0; k0 < K; k0 += 32){
    v8h alo = *(const v8h*)(Arow + k0);        // K = k0+g*16 .. +7
    v8h ahi = *(const v8h*)(Arow + k0 + 8);    // K = k0+g*16+8 .. +15
    v16h a;
    #pragma unroll
    for (int e=0;e<8;e++){ a[e]=alo[e]; a[8+e]=ahi[e]; }
    #pragma unroll
    for (int t=0;t<NSUB;t++){
      v16h bf = *(const v16h*)(Wb + (size_t)t*16*K + k0);     // K = k0+g*16 .. +15
      acc[t] = __builtin_amdgcn_wmma_f32_16x16x32_f16(
                 false, a, false, bf, (short)0, acc[t], false, false);
    }
  }
  #pragma unroll
  for (int t=0;t<NSUB;t++){
    int n = n0 + t*16 + l16;
    #pragma unroll
    for (int r=0;r<8;r++){
      int row = (mt<<4) + g*8 + r;             // C/D layout: VGPR r, half-wave g
      float v = acc[t][r];
      C[(size_t)row*N + n] = v;
      if (WRITE_H) Ch[(size_t)row*N + n] = (_Float16)v;
    }
  }
}

// ---------------- depthwise causal conv (k=4) + SiLU ----------------
__global__ void conv_silu_kernel(const float* __restrict__ xz, const float* __restrict__ cw,
                                 const float* __restrict__ cb, float* __restrict__ xcs,
                                 _Float16* __restrict__ xch){
  int i = blockIdx.x*blockDim.x + threadIdx.x;
  if (i >= BL*DINNER) return;
  int d = i % DINNER;
  int m = i / DINNER;           // b*SEQL + l
  int l = m % SEQL;
  const float* col = xz + (size_t)m*(2*DINNER) + d;
  const float* w   = cw + d*DCONV;
  float acc = cb[d];
  #pragma unroll
  for (int j=0;j<DCONV;j++){
    int lp = l - (DCONV-1) + j;
    if (lp >= 0) acc += w[j] * col[(ptrdiff_t)(j-(DCONV-1))*(2*DINNER)];
  }
  float s = acc * sigmoidf_(acc);
  xcs[i] = s;
  xch[i] = (_Float16)s;
}

// ---------------- dt_proj (K=16) + softplus ----------------
__global__ void delta_kernel(const float* __restrict__ xdbl, const float* __restrict__ dtw,
                             const float* __restrict__ dtb, float* __restrict__ delta){
  int i = blockIdx.x*blockDim.x + threadIdx.x;
  if (i >= BL*DINNER) return;
  int d = i % DINNER;
  int m = i / DINNER;
  const float* row = xdbl + (size_t)m*XDBLW;
  const float* w   = dtw + d*DTRANK;
  float acc = dtb[d];
  #pragma unroll
  for (int r=0;r<DTRANK;r++) acc += row[r]*w[r];
  delta[i] = (acc > 20.f) ? acc : __logf(1.f + __expf(acc));
}

// ================= chunked selective scan (exact 3-phase decomposition) ==========
// thread idx = c*NSTATES + ch*16 + s   (ch = b*DINNER + d)
// Phase 1: per-chunk local scan with h_in = 0 -> hloc, and chunk decay P = exp(Af*sum(delta))
__global__ void scan_chunk_kernel(const float* __restrict__ xcs, const float* __restrict__ delta,
                                  const float* __restrict__ xdbl, const float* __restrict__ Alog,
                                  float* __restrict__ hloc, float* __restrict__ pfac){
  int tid = blockIdx.x*blockDim.x + threadIdx.x;
  if (tid >= NCH*NSTATES) return;
  int s    = tid & 15;
  int rest = tid >> 4;
  int ch   = rest & (BATCH*DINNER - 1);
  int c    = rest >> 11;                 // BATCH*DINNER = 2048 = 2^11
  int d    = ch & (DINNER-1);
  int b    = ch >> 9;
  float Af = -__expf(Alog[d*DSTATE + s]);
  float h  = 0.f, sumd = 0.f;
  size_t mbase = (size_t)b*SEQL + (size_t)c*CLEN;
  for (int j=0;j<CLEN;j++){
    size_t m = mbase + j;
    float dlt = delta[m*DINNER + d];
    float xv  = xcs[m*DINNER + d];
    float Bv  = xdbl[m*XDBLW + DTRANK + s];
    h = __expf(dlt*Af)*h + (dlt*xv)*Bv;
    sumd += dlt;
  }
  hloc[tid] = h;                        // [c][state] layout: coalesced
  pfac[tid] = __expf(Af*sumd);
}

// Phase 2: sequential combine across chunks per state -> exclusive h_in per chunk
__global__ void scan_carry_kernel(const float* __restrict__ hloc, const float* __restrict__ pfac,
                                  float* __restrict__ hin){
  int t = blockIdx.x*blockDim.x + threadIdx.x;
  if (t >= NSTATES) return;
  float h = 0.f;
  for (int c=0;c<NCH;c++){
    size_t idx = (size_t)c*NSTATES + t;
    hin[idx] = h;
    h = pfac[idx]*h + hloc[idx];
  }
}

// Phase 3: replay each chunk from true h_in, reduce over 16 states, write y
__global__ void scan_apply_kernel(const float* __restrict__ xcs, const float* __restrict__ delta,
                                  const float* __restrict__ xdbl, const float* __restrict__ Alog,
                                  const float* __restrict__ Dv, const float* __restrict__ hin,
                                  float* __restrict__ y){
  int tid = blockIdx.x*blockDim.x + threadIdx.x;
  if (tid >= NCH*NSTATES) return;
  int s    = tid & 15;
  int rest = tid >> 4;
  int ch   = rest & (BATCH*DINNER - 1);
  int c    = rest >> 11;
  int d    = ch & (DINNER-1);
  int b    = ch >> 9;
  float Af = -__expf(Alog[d*DSTATE + s]);
  float Dd = Dv[d];
  float h  = hin[tid];
  size_t mbase = (size_t)b*SEQL + (size_t)c*CLEN;
  for (int j=0;j<CLEN;j++){
    size_t m = mbase + j;
    float dlt = delta[m*DINNER + d];
    float xv  = xcs[m*DINNER + d];
    float Bv  = xdbl[m*XDBLW + DTRANK + s];
    float Cv  = xdbl[m*XDBLW + DTRANK + DSTATE + s];
    h = __expf(dlt*Af)*h + (dlt*xv)*Bv;
    float cc = h*Cv;
    cc += __shfl_xor(cc, 1, 32);
    cc += __shfl_xor(cc, 2, 32);
    cc += __shfl_xor(cc, 4, 32);
    cc += __shfl_xor(cc, 8, 32);
    if (s == 0) y[m*DINNER + d] = cc + xv*Dd;
  }
}

// ---------------- gating: g = y * silu(z), to f16 for out_proj ----------------
__global__ void gate_kernel(const float* __restrict__ y, const float* __restrict__ xz,
                            _Float16* __restrict__ gh){
  int i = blockIdx.x*blockDim.x + threadIdx.x;
  if (i >= BL*DINNER) return;
  int d = i % DINNER;
  int m = i / DINNER;
  float z = xz[(size_t)m*(2*DINNER) + DINNER + d];
  gh[i] = (_Float16)(y[i] * z * sigmoidf_(z));
}

// ---------------- head: one wave per (b, node, horizon) ----------------
__global__ void head_kernel(const float* __restrict__ seq, const float* __restrict__ lw,
                            const float* __restrict__ lb, float* __restrict__ out){
  int wave = blockIdx.x*(blockDim.x>>5) + (threadIdx.x>>5);
  int lane = threadIdx.x & 31;
  const int NOUT = BATCH*NNODE*PREDLEN;
  if (wave >= NOUT) return;
  int pr = wave % PREDLEN;
  int n  = (wave / PREDLEN) % NNODE;
  int b  = wave / (PREDLEN*NNODE);
  const float* srow = seq + (size_t)b*SEQL*DMODEL + (size_t)n*(PNUM*DMODEL);
  const float* wrow = lw  + (size_t)pr*(PNUM*DMODEL);
  float acc = 0.f;
  for (int q = lane; q < PNUM*DMODEL; q += 32) acc += srow[q]*wrow[q];
  acc += __shfl_xor(acc, 1, 32);
  acc += __shfl_xor(acc, 2, 32);
  acc += __shfl_xor(acc, 4, 32);
  acc += __shfl_xor(acc, 8, 32);
  acc += __shfl_xor(acc, 16, 32);
  if (lane == 0) out[((size_t)b*PREDLEN + pr)*NNODE + n] = acc + lb[pr];
}

extern "C" void kernel_launch(void* const* d_in, const int* in_sizes, int n_in,
                              void* d_out, int out_size, void* d_ws, size_t ws_size,
                              hipStream_t stream){
  (void)in_sizes; (void)n_in; (void)out_size; (void)ws_size;
  const float* inputs     = (const float*)d_in[0];
  const float* patch_w    = (const float*)d_in[1];
  const float* patch_b    = (const float*)d_in[2];
  const float* in_proj_w  = (const float*)d_in[3];
  const float* conv_w     = (const float*)d_in[4];
  const float* conv_b     = (const float*)d_in[5];
  const float* x_proj_w   = (const float*)d_in[6];
  const float* dt_proj_w  = (const float*)d_in[7];
  const float* dt_proj_b  = (const float*)d_in[8];
  const float* A_log      = (const float*)d_in[9];
  const float* Dp         = (const float*)d_in[10];
  const float* out_proj_w = (const float*)d_in[11];
  const float* lin_w      = (const float*)d_in[12];
  const float* lin_b      = (const float*)d_in[13];
  float* out = (float*)d_out;

  uint8_t* ws = (uint8_t*)d_ws;
  size_t off = 0;
  auto carve = [&](size_t bytes)->uint8_t*{
    uint8_t* p = ws + off; off += (bytes + 255) & ~(size_t)255; return p;
  };
  float*    seq   = (float*)   carve((size_t)BL*DMODEL*4);
  _Float16* seqh  = (_Float16*)carve((size_t)BL*DMODEL*2);
  float*    xz    = (float*)   carve((size_t)BL*2*DINNER*4);
  float*    xcs   = (float*)   carve((size_t)BL*DINNER*4);
  _Float16* xch   = (_Float16*)carve((size_t)BL*DINNER*2);
  float*    xdbl  = (float*)   carve((size_t)BL*XDBLW*4);
  float*    delta = (float*)   carve((size_t)BL*DINNER*4);
  float*    yv    = (float*)   carve((size_t)BL*DINNER*4);
  _Float16* gh    = (_Float16*)carve((size_t)BL*DINNER*2);
  float*    hloc  = (float*)   carve((size_t)NCH*NSTATES*4);
  float*    pfac  = (float*)   carve((size_t)NCH*NSTATES*4);
  float*    hin   = (float*)   carve((size_t)NCH*NSTATES*4);
  _Float16* inw_h = (_Float16*)carve((size_t)NLAYERS*2*DINNER*DMODEL*2);
  _Float16* xpw_h = (_Float16*)carve((size_t)NLAYERS*XDBLW*DINNER*2);
  _Float16* ow_h  = (_Float16*)carve((size_t)NLAYERS*DMODEL*DINNER*2);

  // weight conversion to f16 (once)
  { int n = NLAYERS*2*DINNER*DMODEL;
    f32_to_f16_kernel<<<(n+255)/256,256,0,stream>>>(in_proj_w, inw_h, n); }
  { int n = NLAYERS*XDBLW*DINNER;
    f32_to_f16_kernel<<<(n+255)/256,256,0,stream>>>(x_proj_w, xpw_h, n); }
  { int n = NLAYERS*DMODEL*DINNER;
    f32_to_f16_kernel<<<(n+255)/256,256,0,stream>>>(out_proj_w, ow_h, n); }

  { int n = BATCH*SEQL*DMODEL;
    patch_embed_kernel<<<(n+255)/256,256,0,stream>>>(inputs, patch_w, patch_b, seq, seqh); }

  for (int layer = 0; layer < NLAYERS; layer++){
    const _Float16* inw = inw_h + (size_t)layer*2*DINNER*DMODEL;
    const _Float16* xpw = xpw_h + (size_t)layer*XDBLW*DINNER;
    const _Float16* oww = ow_h  + (size_t)layer*DMODEL*DINNER;
    const float*    alog = A_log + (size_t)layer*DINNER*DSTATE;

    { // in_proj: M=BL, N=1024, K=256
      int M=BL, N=2*DINNER;
      int waves = (M>>4)*(N/64);
      gemm_wmma_f16<DMODEL,4,false><<<(waves+7)/8,256,0,stream>>>(seqh, inw, xz, nullptr, M,N); }

    { int n = BL*DINNER;
      conv_silu_kernel<<<(n+255)/256,256,0,stream>>>(xz, conv_w + (size_t)layer*DINNER*DCONV,
                                                     conv_b + (size_t)layer*DINNER, xcs, xch); }

    { // x_proj: M=BL, N=48, K=512
      int M=BL, N=XDBLW;
      int waves = (M>>4)*(N/48);
      gemm_wmma_f16<DINNER,3,false><<<(waves+7)/8,256,0,stream>>>(xch, xpw, xdbl, nullptr, M,N); }

    { int n = BL*DINNER;
      delta_kernel<<<(n+255)/256,256,0,stream>>>(xdbl, dt_proj_w + (size_t)layer*DINNER*DTRANK,
                                                 dt_proj_b + (size_t)layer*DINNER, delta); }

    { // chunked selective scan
      int n1 = NCH*NSTATES;
      scan_chunk_kernel<<<(n1+255)/256,256,0,stream>>>(xcs, delta, xdbl, alog, hloc, pfac);
      scan_carry_kernel<<<(NSTATES+255)/256,256,0,stream>>>(hloc, pfac, hin);
      scan_apply_kernel<<<(n1+255)/256,256,0,stream>>>(xcs, delta, xdbl, alog,
                                                       Dp + (size_t)layer*DINNER, hin, yv); }

    { int n = BL*DINNER;
      gate_kernel<<<(n+255)/256,256,0,stream>>>(yv, xz, gh); }

    { // out_proj: M=BL, N=256, K=512; also emit f16 seq for next layer
      int M=BL, N=DMODEL;
      int waves = (M>>4)*(N/64);
      gemm_wmma_f16<DINNER,4,true><<<(waves+7)/8,256,0,stream>>>(gh, oww, seq, seqh, M,N); }
  }

  { int waves = BATCH*NNODE*PREDLEN;
    head_kernel<<<(waves+7)/8,256,0,stream>>>(seq, lin_w, lin_b, out); }
}
